// CausalSelfAttention_90872918048897
// MI455X (gfx1250) — compile-verified, matrix operand order checked by reading, not run on hardware
//
#include <hip/hip_runtime.h>

#define BB 2
#define TT 2048
#define CC 1024
#define HH 16
#define DH 64

typedef __attribute__((ext_vector_type(8)))  float          v8f;
typedef __attribute__((ext_vector_type(16))) __bf16         v16bf;
typedef __attribute__((ext_vector_type(2)))  __bf16         bf16x2;
typedef __attribute__((ext_vector_type(16))) unsigned short u16x16;
typedef __attribute__((ext_vector_type(8)))  unsigned short u16x8;

static __device__ __forceinline__ unsigned short f2bf(float f) {
#if __has_builtin(__builtin_amdgcn_cvt_pk_bf16_f32)
  bf16x2 p = __builtin_amdgcn_cvt_pk_bf16_f32(f, 0.0f);
  return (unsigned short)(__builtin_bit_cast(unsigned, p) & 0xFFFFu);
#else
  unsigned u = __builtin_bit_cast(unsigned, f);
  u += 0x7FFFu + ((u >> 16) & 1u);          // round-to-nearest-even
  return (unsigned short)(u >> 16);
#endif
}

static __device__ __forceinline__ v8f wmma_bf16(v16bf a, v16bf b, v8f c) {
  // (neg_a, A, neg_b, B, c_mod, C, reuse_a, reuse_b)
  return __builtin_amdgcn_wmma_f32_16x16x32_bf16(false, a, false, b, (short)0, c, false, false);
}

// 16x32 bf16 WMMA operand fragment from a row-major bf16 matrix (ld elements).
// ISA 16-bit operand layout: lane m(0..15)/m+16 holds row m; per lane two
// contiguous 8-element chunks at k0+(lane>>4)*8 and k0+16+(lane>>4)*8.
// Works for global (global_load_b128) and LDS (ds_load_b128) pointers.
static __device__ __forceinline__ v16bf load_bf_frag(const unsigned short* base,
                                                     int ld, int row0, int k0, int lane) {
  int r  = row0 + (lane & 15);
  int kc = k0 + ((lane >> 4) << 3);
  const unsigned short* rp = base + (size_t)r * ld + kc;
  u16x8 c0 = *(const u16x8*)(rp);        // 16B load
  u16x8 c1 = *(const u16x8*)(rp + 16);   // 16B load
  u16x16 t;
#pragma unroll
  for (int i = 0; i < 8; ++i) { t[i] = c0[i]; t[i + 8] = c1[i]; }
  return __builtin_bit_cast(v16bf, t);
}

// CDNA5 async copy: 16B global -> LDS, tracked by ASYNCcnt (CDNA5 ISA).
static __device__ __forceinline__ void async_g2l_b128(unsigned lds_byte_off,
                                                      const void* gptr) {
  unsigned long long ga = (unsigned long long)(size_t)gptr;
  asm volatile("global_load_async_to_lds_b128 %0, %1, off"
               :: "v"(lds_byte_off), "v"(ga) : "memory");
}
static __device__ __forceinline__ void wait_async0() {
  asm volatile("s_wait_asynccnt 0" ::: "memory");
}

// ----------------------------------------------------- one-time bf16 convert
// [wq|wk|wv|wo] (4*2^20) then x (4*2^20)
__global__ __launch_bounds__(256) void cvt_kernel(
    const float* __restrict__ wq, const float* __restrict__ wk,
    const float* __restrict__ wv, const float* __restrict__ wo,
    const float* __restrict__ x,
    unsigned short* __restrict__ wbf, unsigned short* __restrict__ xb) {
  int i = blockIdx.x * 256 + threadIdx.x;        // 8 * 2^20 threads exactly
  const int WTOT = 4 << 20;
  if (i < WTOT) {
    int mat = i >> 20;
    int off = i & ((1 << 20) - 1);
    const float* src = (mat == 0) ? wq : (mat == 1) ? wk : (mat == 2) ? wv : wo;
    wbf[i] = f2bf(src[off]);
  } else {
    int j = i - WTOT;
    xb[j] = f2bf(x[j]);
  }
}

// --------------------------------------------------------------- fused QKV
// out = x @ W^T + pos_emb. Per wave: 16(t) x 64(c_out) strip, 4 accumulators,
// A-fragment reused across 4 B tiles. Q,K stored [B,H,T,Dh]; V as [B,H,Dh,T].
__global__ __launch_bounds__(256) void qkv_kernel(
    const unsigned short* __restrict__ xb, const float* __restrict__ pe,
    const unsigned short* __restrict__ wbf,
    unsigned short* __restrict__ qb, unsigned short* __restrict__ kb,
    unsigned short* __restrict__ vtb) {
  int lane = threadIdx.x & 31;
  int wid  = threadIdx.x >> 5;
  int gw   = blockIdx.x * 8 + wid;               // 12288 waves
  int mat  = gw >> 12;                           // 0=Q 1=K 2=V
  int rem  = gw & 4095;
  int rowbase = (rem >> 4) << 4;                 // over B*T
  int colbase = (rem & 15) << 6;                 // over C, 64-wide strip
  const unsigned short* W = wbf + (size_t)mat * (CC * CC);

  int mh = (lane >> 4) << 3;
  int nl = lane & 15;

  v8f acc[4];
#pragma unroll
  for (int j = 0; j < 4; ++j)
#pragma unroll
    for (int r = 0; r < 8; ++r) {                // C init = pos_emb (f32 exact)
      int t = (rowbase + r + mh) & (TT - 1);
      acc[j][r] = pe[(size_t)t * CC + colbase + j * 16 + nl];
    }
  for (int k = 0; k < CC; k += 32) {
    // gfx1250 prefetch (global_prefetch_b8) of the next weight chunk
    __builtin_prefetch(W + (size_t)(colbase + nl) * CC + k + 128, 0, 1);
    v16bf a = load_bf_frag(xb, CC, rowbase, k, lane);
#pragma unroll
    for (int j = 0; j < 4; ++j) {
      v16bf b = load_bf_frag(W, CC, colbase + j * 16, k, lane);
      acc[j] = wmma_bf16(a, b, acc[j]);
    }
  }
#pragma unroll
  for (int j = 0; j < 4; ++j)
#pragma unroll
    for (int r = 0; r < 8; ++r) {
      int m  = rowbase + r + mh;
      int bb = m >> 11;                          // /T
      int t  = m & (TT - 1);
      int co = colbase + j * 16 + nl;
      int h  = co >> 6, d = co & 63;
      unsigned short v = f2bf(acc[j][r]);
      size_t bh = (size_t)bb * HH + h;
      if (mat == 0)      qb[(bh * TT + t) * DH + d] = v;
      else if (mat == 1) kb[(bh * TT + t) * DH + d] = v;
      else               vtb[(bh * DH + d) * TT + t] = v;
    }
}

// ------------------------------------------------------------ flash attention
// Block = 8 waves = 8 query tiles of one (b,h). K/V 32-key tiles are staged
// into double-buffered LDS with async global->LDS copies: tile s+1 is issued
// before computing tile s, so the DMA overlaps the 8 WMMAs + softmax; the
// s_wait_asynccnt 0 + barrier at the end of each step covers both the RAW
// (next tile staged) and WAR (buffer re-use two steps later) hazards.
__global__ __launch_bounds__(256) void attn_kernel(
    const unsigned short* __restrict__ qb, const unsigned short* __restrict__ kb,
    const unsigned short* __restrict__ vtb, unsigned short* __restrict__ yb) {
  __shared__ __align__(16) unsigned short ldsK[2][32 * 64]; // keys x dh, 2x4KB
  __shared__ __align__(16) unsigned short ldsV[2][64 * 32]; // dh x keys, 2x4KB
  __shared__ __align__(16) unsigned short plds[8][16 * 32]; // P bounce, 8KB

  int lane = threadIdx.x & 31;
  int wid  = threadIdx.x >> 5;
  int bh     = blockIdx.x >> 4;                  // 32 (b,h) pairs
  int qgroup = blockIdx.x & 15;
  int qt     = qgroup * 8 + wid;                 // 0..127

  const unsigned short* qbase = qb + (size_t)bh * TT * DH;
  const unsigned short* kbase = kb + (size_t)bh * TT * DH;
  const unsigned short* vbase = vtb + (size_t)bh * DH * TT;

  v16bf q0 = load_bf_frag(qbase, DH, qt * 16, 0, lane);
  v16bf q1 = load_bf_frag(qbase, DH, qt * 16, 32, lane);

  v8f z = {0.f, 0.f, 0.f, 0.f, 0.f, 0.f, 0.f, 0.f};
  v8f o0 = z, o1 = z, o2 = z, o3 = z;
  float mrow[8], lrow[8];
#pragma unroll
  for (int r = 0; r < 8; ++r) { mrow[r] = -1e30f; lrow[r] = 0.f; }

  int mh = (lane >> 4) << 3;
  int nl = lane & 15;
  int qrowbase = qt * 16 + mh;
  const float scale = 0.03125f;                  // 1/sqrt(1024)
  int my_steps  = (((qt + 1) * 16) + 31) >> 5;
  int max_steps = (((qgroup * 8 + 7 + 1) * 16) + 31) >> 5;

  // cooperative staging: 256 threads x one 16B chunk for each of K,V
  int c    = threadIdx.x;
  int kkey = c >> 3, kdh = (c & 7) << 3;         // K: 32 rows x 64 cols
  int vdh  = c >> 2, vkc = (c & 3) << 3;         // V: 64 rows x 32 cols
  unsigned kOff[2], vOff[2];
#pragma unroll
  for (int bsel = 0; bsel < 2; ++bsel) {
    kOff[bsel] = (unsigned)(size_t)&ldsK[bsel][kkey * 64 + kdh];
    vOff[bsel] = (unsigned)(size_t)&ldsV[bsel][vdh * 32 + vkc];
  }

  unsigned short* pl = plds[wid];

  // prologue: stage tile 0 into buffer 0
  async_g2l_b128(kOff[0], kbase + (size_t)kkey * DH + kdh);
  async_g2l_b128(vOff[0], vbase + (size_t)vdh * TT + vkc);
  wait_async0();
  __syncthreads();

  for (int s = 0; s < max_steps; ++s) {
    int k0  = s << 5;
    int cur = s & 1;
    if (s + 1 < max_steps) {                     // issue next tile, no wait yet
      int kn = (s + 1) << 5, nxt = cur ^ 1;
      async_g2l_b128(kOff[nxt], kbase + (size_t)(kn + kkey) * DH + kdh);
      async_g2l_b128(vOff[nxt], vbase + (size_t)vdh * TT + kn + vkc);
    }

    if (s < my_steps) {
      const unsigned short* lk = ldsK[cur];
      const unsigned short* lv = ldsV[cur];
      // S = Q K^T  (two 16-key sub-tiles, Dh=64 -> 2 WMMAs each)
      v8f s0 = wmma_bf16(q0, load_bf_frag(lk, DH, 0, 0, lane), z);
      s0     = wmma_bf16(q1, load_bf_frag(lk, DH, 0, 32, lane), s0);
      v8f s1 = wmma_bf16(q0, load_bf_frag(lk, DH, 16, 0, lane), z);
      s1     = wmma_bf16(q1, load_bf_frag(lk, DH, 16, 32, lane), s1);

      int key0 = k0 + nl, key1 = k0 + 16 + nl;
      float p0[8], p1[8];
#pragma unroll
      for (int r = 0; r < 8; ++r) {
        int qr  = qrowbase + r;
        float a = (key0 <= qr) ? s0[r] * scale : -1e30f;
        float b = (key1 <= qr) ? s1[r] * scale : -1e30f;
        float mx = fmaxf(a, b);
        mx = fmaxf(mx, __shfl_xor(mx, 1));
        mx = fmaxf(mx, __shfl_xor(mx, 2));
        mx = fmaxf(mx, __shfl_xor(mx, 4));
        mx = fmaxf(mx, __shfl_xor(mx, 8));       // row max within 16-lane half
        float mnew  = fmaxf(mrow[r], mx);
        float alpha = __expf(mrow[r] - mnew);
        float e0 = __expf(a - mnew);
        float e1 = __expf(b - mnew);
        float rs = e0 + e1;
        rs += __shfl_xor(rs, 1);
        rs += __shfl_xor(rs, 2);
        rs += __shfl_xor(rs, 4);
        rs += __shfl_xor(rs, 8);
        lrow[r] = lrow[r] * alpha + rs;
        mrow[r] = mnew;
        p0[r] = e0; p1[r] = e1;
        o0[r] *= alpha; o1[r] *= alpha; o2[r] *= alpha; o3[r] *= alpha;
      }
      // C-layout -> A-layout relayout of P through per-wave LDS (DS in-order)
#pragma unroll
      for (int r = 0; r < 8; ++r) {
        int mr = r + mh;
        pl[mr * 32 + nl]      = f2bf(p0[r]);
        pl[mr * 32 + 16 + nl] = f2bf(p1[r]);
      }
      v16bf pf;
      {
        int kc = (lane >> 4) << 3;
        u16x8 c0 = *(const u16x8*)(pl + nl * 32 + kc);
        u16x8 c1 = *(const u16x8*)(pl + nl * 32 + 16 + kc);
        u16x16 t;
#pragma unroll
        for (int i = 0; i < 8; ++i) { t[i] = c0[i]; t[i + 8] = c1[i]; }
        pf = __builtin_bit_cast(v16bf, t);
      }
      // O += P @ V   (V tile in LDS as [dh][keys], lane = dh column)
      o0 = wmma_bf16(pf, load_bf_frag(lv, 32, 0,  0, lane), o0);
      o1 = wmma_bf16(pf, load_bf_frag(lv, 32, 16, 0, lane), o1);
      o2 = wmma_bf16(pf, load_bf_frag(lv, 32, 32, 0, lane), o2);
      o3 = wmma_bf16(pf, load_bf_frag(lv, 32, 48, 0, lane), o3);
    }

    wait_async0();                               // next tile landed (overlapped)
    __syncthreads();                             // visible to all waves; WAR safe
  }

  int b = bh >> 4, h = bh & 15;
#pragma unroll
  for (int r = 0; r < 8; ++r) {
    float inv = 1.0f / lrow[r];
    int t = qt * 16 + r + mh;
    unsigned short* yrow = yb + ((size_t)b * TT + t) * CC + h * 64;
    yrow[nl]      = f2bf(o0[r] * inv);
    yrow[16 + nl] = f2bf(o1[r] * inv);
    yrow[32 + nl] = f2bf(o2[r] * inv);
    yrow[48 + nl] = f2bf(o3[r] * inv);
  }
}

// ---------------------------------------------------------- output projection
__global__ __launch_bounds__(256) void outproj_kernel(
    const unsigned short* __restrict__ yb, const unsigned short* __restrict__ wo,
    float* __restrict__ out) {
  int lane = threadIdx.x & 31;
  int wid  = threadIdx.x >> 5;
  int gw   = blockIdx.x * 8 + wid;               // 4096 waves
  int rowbase = (gw >> 4) << 4;
  int colbase = (gw & 15) << 6;
  int mh = (lane >> 4) << 3;
  int nl = lane & 15;

  v8f acc[4];
#pragma unroll
  for (int j = 0; j < 4; ++j) acc[j] = (v8f){0.f,0.f,0.f,0.f,0.f,0.f,0.f,0.f};
  for (int k = 0; k < CC; k += 32) {
    __builtin_prefetch(wo + (size_t)(colbase + nl) * CC + k + 128, 0, 1);
    v16bf a = load_bf_frag(yb, CC, rowbase, k, lane);
#pragma unroll
    for (int j = 0; j < 4; ++j) {
      v16bf b = load_bf_frag(wo, CC, colbase + j * 16, k, lane);
      acc[j] = wmma_bf16(a, b, acc[j]);
    }
  }
#pragma unroll
  for (int j = 0; j < 4; ++j)
#pragma unroll
    for (int r = 0; r < 8; ++r) {
      int m = rowbase + r + mh;
      out[(size_t)m * CC + colbase + j * 16 + nl] = acc[j][r];
    }
}

extern "C" void kernel_launch(void* const* d_in, const int* in_sizes, int n_in,
                              void* d_out, int out_size, void* d_ws, size_t ws_size,
                              hipStream_t stream) {
  const float* x  = (const float*)d_in[0];
  const float* pe = (const float*)d_in[1];
  const float* wq = (const float*)d_in[2];
  const float* wk = (const float*)d_in[3];
  const float* wv = (const float*)d_in[4];
  const float* wo = (const float*)d_in[5];

  char* ws = (char*)d_ws;
  // [wq|wk|wv|wo bf16 8MB][x bf16 8MB][Q 8MB][K 8MB][Vt 8MB][Y 8MB] = 48MB
  unsigned short* wbf = (unsigned short*)(ws);
  unsigned short* xb  = (unsigned short*)(ws + (size_t)8  * (1 << 20));
  unsigned short* qbf = (unsigned short*)(ws + (size_t)16 * (1 << 20));
  unsigned short* kbf = (unsigned short*)(ws + (size_t)24 * (1 << 20));
  unsigned short* vtb = (unsigned short*)(ws + (size_t)32 * (1 << 20));
  unsigned short* yb  = (unsigned short*)(ws + (size_t)40 * (1 << 20));

  cvt_kernel<<<dim3(32768), dim3(256), 0, stream>>>(wq, wk, wv, wo, x, wbf, xb);
  qkv_kernel<<<dim3(1536), dim3(256), 0, stream>>>(xb, pe, wbf, qbf, kbf, vtb);
  attn_kernel<<<dim3(512), dim3(256), 0, stream>>>(qbf, kbf, vtb, yb);
  outproj_kernel<<<dim3(512), dim3(256), 0, stream>>>(yb, wbf + (size_t)3 * CC * CC,
                                                      (float*)d_out);
}
